// VQVAE_part_no_decoder_21414706938507
// MI455X (gfx1250) — matrix operand order repaired and checked by reading, not compiled, not run on hardware
//
#include <hip/hip_runtime.h>
#include <hip/hip_bf16.h>

typedef __bf16 bf16_t;
typedef __attribute__((ext_vector_type(16))) __bf16 v16bf;
typedef __attribute__((ext_vector_type(8)))  float  v8f;
typedef __attribute__((ext_vector_type(4)))  int    v4i;

#define NIMG     32
#define WIDTH    512
#define CODE_DIM 512
#define NB_CODE  1024
#define TQ       256
#define NROWS    (NIMG * TQ)          // 8192 quantized rows
#define CIN_PAD  288                  // 263 padded to multiple of 32

union Frag16 { v16bf v; uint4 q[2]; unsigned short u[16]; };

#if __has_builtin(__builtin_amdgcn_global_load_async_to_lds_b128)
#define HAVE_ASYNC_LDS 1
typedef __attribute__((address_space(1))) v4i* as1_v4i;
typedef __attribute__((address_space(3))) v4i* as3_v4i;
#else
#define HAVE_ASYNC_LDS 0
#endif

// copy 16B global -> LDS (async CDNA5 path when available)
__device__ __forceinline__ void g2l_b128(const void* g, void* l) {
#if HAVE_ASYNC_LDS
    __builtin_amdgcn_global_load_async_to_lds_b128((as1_v4i)(g), (as3_v4i)(l), 0, 0);
#else
    *(uint4*)l = *(const uint4*)g;
#endif
}

__device__ __forceinline__ void async_wait_all() {
#if HAVE_ASYNC_LDS
#if __has_builtin(__builtin_amdgcn_s_wait_asynccnt)
    __builtin_amdgcn_s_wait_asynccnt(0);
#else
    asm volatile("s_wait_asynccnt 0x0" ::: "memory");
#endif
#endif
}

// ---------------------------------------------------------------------------
// Conv1d as GEMM with WMMA bf16.  4-wave block computes 64(M) x 64(N) tile.
// in : (N, T_in, Cin) bf16, Cin % 32 == 0
// wT : (Cout, Kw*Cin) bf16  (wT[o][tap*Cin + c] = w[o][c][tap])
// out: (N, T_out, Cout) bf16
// Shared 32x64 B tile is staged into LDS (double-buffered) with async
// global->LDS copies; each wave reads B fragments via ds_load_b128.
// A layout (16-bit 16x32): lanes 0-15 row M=lane, K = kh*16 + 0..15 contiguous;
// B: lane holds one column, K = kh*16 + 0..15 contiguous.
// ---------------------------------------------------------------------------
__global__ __launch_bounds__(128) void conv_wmma_kernel(
    const bf16_t* __restrict__ in, const bf16_t* __restrict__ wT,
    const float* __restrict__ bias, const bf16_t* __restrict__ res,
    bf16_t* __restrict__ out,
    int T_in, int T_out, int Cin, int Cout,
    int Kw, int stride, int pad, int dil,
    int pre_relu, int post_relu)
{
    __shared__ __align__(16) unsigned char sB[2][64 * 64];   // 2 x 4KB: 64 cols x 2 khalf x 32B

    const int tid   = threadIdx.x;
    const int wave  = tid >> 5;
    const int lane  = tid & 31;
    const int mtile = blockIdx.x * 4 + wave;
    const int nt4   = blockIdx.y;
    const int Kg    = Kw * Cin;
    const int nk    = Kg / 32;
    const int m     = lane & 15;
    const int khalf = lane >> 4;

    const int grow_a = mtile * 16 + m;          // A row = n*T_out + t
    const int n_img  = grow_a / T_out;
    const int t_out  = grow_a - n_img * T_out;
    const int col    = nt4 * 64 + m;            // output channel (per 16-tile)

    // B staging assignment for this thread: column j (0..63), half h (0..1)
    const int sj = tid >> 1;
    const int sh = tid & 1;
    const bf16_t* sB_src = wT + (size_t)(nt4 * 64 + sj) * Kg + sh * 16;
    unsigned char* sB_dst0 = &sB[0][(sj * 2 + sh) * 32];
    unsigned char* sB_dst1 = &sB[1][(sj * 2 + sh) * 32];

    v8f acc0 = {}, acc1 = {}, acc2 = {}, acc3 = {};

    // prologue: stage K-step 0 into buffer 0
    g2l_b128(sB_src, sB_dst0);
    g2l_b128(sB_src + 8, sB_dst0 + 16);

    for (int ks = 0; ks < nk; ++ks) {
        const int g = ks * 32;
        unsigned char* bufc = (ks & 1) ? &sB[1][0] : &sB[0][0];

        async_wait_all();
        __syncthreads();                        // staged B visible to all waves

        if (ks + 1 < nk) {                      // overlap next stage with compute
            unsigned char* dstn = ((ks + 1) & 1) ? sB_dst1 : sB_dst0;
            const bf16_t* srcn = sB_src + (ks + 1) * 32;
            g2l_b128(srcn, dstn);
            g2l_b128(srcn + 8, dstn + 16);
        }

        // ---- A fragment (direct global, per wave) ----
        const int tap  = g / Cin;
        const int cb0  = g - tap * Cin;         // 32-block stays inside one tap (Cin%32==0)
        const int t_in = t_out * stride + tap * dil - pad;

        Frag16 A;
        if ((unsigned)t_in < (unsigned)T_in) {
            const bf16_t* p = in + ((size_t)n_img * T_in + t_in) * Cin + cb0 + khalf * 16;
            A.q[0] = *(const uint4*)(p);
            A.q[1] = *(const uint4*)(p + 8);
            if (pre_relu) {
                #pragma unroll
                for (int j = 0; j < 16; ++j)
                    if (A.u[j] & 0x8000u) A.u[j] = 0;   // bf16 relu via sign bit
            }
        } else {
            A.q[0] = make_uint4(0,0,0,0);
            A.q[1] = make_uint4(0,0,0,0);
        }

        // ---- B fragments from LDS ----
        Frag16 B0, B1, B2, B3;
        {
            const unsigned char* b0 = bufc + ((m)      * 2 + khalf) * 32;
            const unsigned char* b1 = bufc + ((m + 16) * 2 + khalf) * 32;
            const unsigned char* b2 = bufc + ((m + 32) * 2 + khalf) * 32;
            const unsigned char* b3 = bufc + ((m + 48) * 2 + khalf) * 32;
            B0.q[0] = *(const uint4*)(b0); B0.q[1] = *(const uint4*)(b0 + 16);
            B1.q[0] = *(const uint4*)(b1); B1.q[1] = *(const uint4*)(b1 + 16);
            B2.q[0] = *(const uint4*)(b2); B2.q[1] = *(const uint4*)(b2 + 16);
            B3.q[0] = *(const uint4*)(b3); B3.q[1] = *(const uint4*)(b3 + 16);
        }

        acc0 = __builtin_amdgcn_wmma_f32_16x16x32_bf16(false, A.v, false, B0.v, (short)0, acc0, false, false);
        acc1 = __builtin_amdgcn_wmma_f32_16x16x32_bf16(false, A.v, false, B1.v, (short)0, acc1, false, false);
        acc2 = __builtin_amdgcn_wmma_f32_16x16x32_bf16(false, A.v, false, B2.v, (short)0, acc2, false, false);
        acc3 = __builtin_amdgcn_wmma_f32_16x16x32_bf16(false, A.v, false, B3.v, (short)0, acc3, false, false);

        __syncthreads();   // all waves done reading this buffer before it is restaged
    }

    // D layout: VGPR r -> (M = r or 8+r, N = lane&15)
    const int rowBase = (lane < 16) ? 0 : 8;
    const float b0 = bias[col], b1 = bias[col + 16], b2 = bias[col + 32], b3 = bias[col + 48];

    #pragma unroll
    for (int r = 0; r < 8; ++r) {
        const int grow = mtile * 16 + rowBase + r;
        const size_t ob = (size_t)grow * Cout;
        float v0 = acc0[r] + b0, v1 = acc1[r] + b1, v2 = acc2[r] + b2, v3 = acc3[r] + b3;
        if (res) {
            v0 += (float)res[ob + col];
            v1 += (float)res[ob + col + 16];
            v2 += (float)res[ob + col + 32];
            v3 += (float)res[ob + col + 48];
        }
        if (post_relu) {
            v0 = v0 > 0.f ? v0 : 0.f; v1 = v1 > 0.f ? v1 : 0.f;
            v2 = v2 > 0.f ? v2 : 0.f; v3 = v3 > 0.f ? v3 : 0.f;
        }
        out[ob + col]      = (bf16_t)v0;
        out[ob + col + 16] = (bf16_t)v1;
        out[ob + col + 32] = (bf16_t)v2;
        out[ob + col + 48] = (bf16_t)v3;
    }
}

// ---------------------------------------------------------------------------
// VQ argmin: dist = ||x||^2 - 2 x.c + ||c||^2 ; ||x||^2 constant per row,
// so argmin only needs cnorm[c] - 2*dot.  WMMA GEMM + cross-lane min-reduce.
// ---------------------------------------------------------------------------
__global__ __launch_bounds__(32) void vq_argmin_kernel(
    const bf16_t* __restrict__ e,     // (NROWS, CODE_DIM) bf16
    const bf16_t* __restrict__ cbh,   // (NB_CODE, CODE_DIM) bf16
    const float*  __restrict__ cnorm, // (NB_CODE)
    int* __restrict__ code_idx)       // (NROWS)
{
    const int lane  = threadIdx.x;
    const int mtile = blockIdx.x;     // NROWS/16
    const int m     = lane & 15;
    const int khalf = lane >> 4;

    float bestV[8];
    int   bestI[8];
    #pragma unroll
    for (int r = 0; r < 8; ++r) { bestV[r] = 3.0e38f; bestI[r] = 0; }

    for (int nt = 0; nt < NB_CODE / 16; ++nt) {
        v8f acc = {};
        for (int g = 0; g < CODE_DIM; g += 32) {
            Frag16 A, B;
            const bf16_t* ap = e + ((size_t)(mtile * 16 + m)) * CODE_DIM + g + khalf * 16;
            A.q[0] = *(const uint4*)(ap);
            A.q[1] = *(const uint4*)(ap + 8);
            const bf16_t* bp = cbh + ((size_t)(nt * 16 + m)) * CODE_DIM + g + khalf * 16;
            B.q[0] = *(const uint4*)(bp);
            B.q[1] = *(const uint4*)(bp + 8);
            acc = __builtin_amdgcn_wmma_f32_16x16x32_bf16(false, A.v, false, B.v, (short)0, acc, false, false);
        }
        const int cidx = nt * 16 + m;
        const float cn = cnorm[cidx];
        #pragma unroll
        for (int r = 0; r < 8; ++r) {
            float s = cn - 2.0f * acc[r];
            if (s < bestV[r]) { bestV[r] = s; bestI[r] = cidx; }
        }
    }

    // min-reduce across the 16 lanes that share each row
    #pragma unroll
    for (int off = 8; off >= 1; off >>= 1) {
        #pragma unroll
        for (int r = 0; r < 8; ++r) {
            float ov = __shfl_xor(bestV[r], off, 16);
            int   oi = __shfl_xor(bestI[r], off, 16);
            if (ov < bestV[r] || (ov == bestV[r] && oi < bestI[r])) { bestV[r] = ov; bestI[r] = oi; }
        }
    }
    if ((lane & 15) == 0) {
        const int rowBase = (lane < 16) ? 0 : 8;
        #pragma unroll
        for (int r = 0; r < 8; ++r)
            code_idx[mtile * 16 + rowBase + r] = bestI[r];
    }
}

// ------------------------------ small kernels ------------------------------
__global__ void pack_x_kernel(const float* __restrict__ x, bf16_t* __restrict__ xp) {
    size_t i = (size_t)blockIdx.x * blockDim.x + threadIdx.x;
    const size_t total = (size_t)NIMG * 2048 * CIN_PAD;
    if (i >= total) return;
    int c = (int)(i % CIN_PAD);
    size_t rt = i / CIN_PAD;
    xp[i] = (c < 263) ? (bf16_t)x[rt * 263 + c] : (bf16_t)0.f;
}

__global__ void pack_w_kernel(const float* __restrict__ w, bf16_t* __restrict__ wT,
                              int O, int I, int Kw, int Cin) {
    size_t i = (size_t)blockIdx.x * blockDim.x + threadIdx.x;
    const int Kg = Kw * Cin;
    if (i >= (size_t)O * Kg) return;
    int o = (int)(i / Kg);
    int rem = (int)(i - (size_t)o * Kg);
    int tap = rem / Cin;
    int c = rem - tap * Cin;
    wT[i] = (c < I) ? (bf16_t)w[((size_t)o * I + c) * Kw + tap] : (bf16_t)0.f;
}

__global__ __launch_bounds__(256) void pack_cb_kernel(const float* __restrict__ cb,
                                                      bf16_t* __restrict__ cbh,
                                                      float* __restrict__ cnorm) {
    __shared__ float red[256];
    const int e = blockIdx.x;       // codebook entry
    const int t = threadIdx.x;
    float s = 0.f;
    for (int c = t; c < CODE_DIM; c += 256) {
        float v = cb[(size_t)e * CODE_DIM + c];
        cbh[(size_t)e * CODE_DIM + c] = (bf16_t)v;
        s += v * v;
    }
    red[t] = s; __syncthreads();
    for (int st = 128; st > 0; st >>= 1) {
        if (t < st) red[t] += red[t + st];
        __syncthreads();
    }
    if (t == 0) cnorm[e] = red[0];
}

__global__ void vq_init_kernel(unsigned* __restrict__ counts, float* __restrict__ loss) {
    int i = blockIdx.x * blockDim.x + threadIdx.x;
    if (i < NB_CODE) counts[i] = 0u;
    if (i == 0) *loss = 0.f;
}

__global__ void vq_count_kernel(const int* __restrict__ idx, unsigned* __restrict__ counts) {
    int i = blockIdx.x * blockDim.x + threadIdx.x;
    if (i < NROWS) atomicAdd(&counts[idx[i]], 1u);
}

__global__ void vq_gather_kernel(const int* __restrict__ idx, const float* __restrict__ cb,
                                 float* __restrict__ xq) {
    // x_q: (N, C, Tq); forward value of straight-through = codebook[idx]
    size_t i = (size_t)blockIdx.x * blockDim.x + threadIdx.x;
    const size_t total = (size_t)NIMG * WIDTH * TQ;
    if (i >= total) return;
    int t = (int)(i & (TQ - 1));
    int c = (int)((i >> 8) & (WIDTH - 1));
    int n = (int)(i >> 17);
    xq[i] = cb[(size_t)idx[n * TQ + t] * CODE_DIM + c];
}

__global__ __launch_bounds__(256) void vq_commit_kernel(const bf16_t* __restrict__ e,
                                                        const float* __restrict__ cb,
                                                        const int* __restrict__ idx,
                                                        float* __restrict__ out_loss) {
    __shared__ float red[256];
    const int row = blockIdx.x;             // NROWS blocks
    const float* cr = cb + (size_t)idx[row] * CODE_DIM;
    const bf16_t* er = e + (size_t)row * CODE_DIM;
    float s = 0.f;
    for (int c = threadIdx.x; c < CODE_DIM; c += 256) {
        float d = (float)er[c] - cr[c];
        s += d * d;
    }
    red[threadIdx.x] = s; __syncthreads();
    for (int st = 128; st > 0; st >>= 1) {
        if (threadIdx.x < st) red[threadIdx.x] += red[threadIdx.x + st];
        __syncthreads();
    }
    if (threadIdx.x == 0)
        atomicAdd(out_loss, red[0] * (1.0f / ((float)NROWS * (float)CODE_DIM)));
}

__global__ __launch_bounds__(1024) void vq_perplexity_kernel(const unsigned* __restrict__ counts,
                                                             float* __restrict__ out) {
    __shared__ float red[1024];
    const int i = threadIdx.x;              // NB_CODE threads
    float p = (float)counts[i] / (float)NROWS;
    red[i] = -p * __logf(p + 1e-7f);
    __syncthreads();
    for (int st = 512; st > 0; st >>= 1) {
        if (i < st) red[i] += red[i + st];
        __syncthreads();
    }
    if (i == 0) *out = __expf(red[0]);
}

// ------------------------------ host driver --------------------------------
static inline void launch_conv(hipStream_t s, const bf16_t* in, const bf16_t* wT,
                               const float* bias, const bf16_t* res, bf16_t* out,
                               int T_in, int T_out, int Cin, int Cout,
                               int Kw, int stride, int pad, int dil,
                               int pre_relu, int post_relu) {
    dim3 grid((NIMG * T_out) / 64, Cout / 64);   // 4 waves/block, 64x64 tile
    conv_wmma_kernel<<<grid, 128, 0, s>>>(in, wT, bias, res, out, T_in, T_out, Cin, Cout,
                                          Kw, stride, pad, dil, pre_relu, post_relu);
}

static inline void launch_pack_w(hipStream_t s, const float* w, bf16_t* wT,
                                 int O, int I, int Kw, int Cin) {
    size_t total = (size_t)O * Kw * Cin;
    pack_w_kernel<<<(unsigned)((total + 255) / 256), 256, 0, s>>>(w, wT, O, I, Kw, Cin);
}

extern "C" void kernel_launch(void* const* d_in, const int* in_sizes, int n_in,
                              void* d_out, int out_size, void* d_ws, size_t ws_size,
                              hipStream_t stream) {
    const float* x          = (const float*)d_in[0];
    const float* conv_in_w  = (const float*)d_in[1];
    const float* conv_in_b  = (const float*)d_in[2];
    const float* down_w     = (const float*)d_in[3];
    const float* down_b     = (const float*)d_in[4];
    const float* res_w1     = (const float*)d_in[5];
    const float* res_b1     = (const float*)d_in[6];
    const float* res_w2     = (const float*)d_in[7];
    const float* res_b2     = (const float*)d_in[8];
    const float* conv_out_w = (const float*)d_in[9];
    const float* conv_out_b = (const float*)d_in[10];
    const float* codebook   = (const float*)d_in[11];

    float* out      = (float*)d_out;
    float* out_loss = out + (size_t)NIMG * WIDTH * TQ;
    float* out_perp = out_loss + 1;

    char* ws = (char*)d_ws;
    size_t off = 0;
    auto alloc = [&](size_t bytes) -> void* {
        void* p = ws + off;
        off += (bytes + 255) & ~(size_t)255;
        return p;
    };

    bf16_t* xp   = (bf16_t*)alloc((size_t)NIMG * 2048 * CIN_PAD * 2);   // 36 MB
    bf16_t* actA = (bf16_t*)alloc((size_t)NIMG * 2048 * WIDTH * 2);     // 64 MB
    bf16_t* actB = (bf16_t*)alloc((size_t)NIMG * 1024 * WIDTH * 2);     // 32 MB
    bf16_t* actC = (bf16_t*)alloc((size_t)NIMG * 1024 * WIDTH * 2);     // 32 MB
    bf16_t* wt_in   = (bf16_t*)alloc((size_t)WIDTH * 3 * CIN_PAD * 2);
    bf16_t* wt_down = (bf16_t*)alloc((size_t)3 * WIDTH * 4 * WIDTH * 2);
    bf16_t* wt_r1   = (bf16_t*)alloc((size_t)9 * WIDTH * 3 * WIDTH * 2);
    bf16_t* wt_r2   = (bf16_t*)alloc((size_t)9 * WIDTH * 1 * WIDTH * 2);
    bf16_t* wt_out  = (bf16_t*)alloc((size_t)CODE_DIM * 3 * WIDTH * 2);
    bf16_t* cbh     = (bf16_t*)alloc((size_t)NB_CODE * CODE_DIM * 2);
    float*  cnorm   = (float*)alloc((size_t)NB_CODE * 4);
    int*    codeidx = (int*)alloc((size_t)NROWS * 4);
    unsigned* counts = (unsigned*)alloc((size_t)NB_CODE * 4);

    // ---- packing ----
    {
        size_t total = (size_t)NIMG * 2048 * CIN_PAD;
        pack_x_kernel<<<(unsigned)((total + 255) / 256), 256, 0, stream>>>(x, xp);
    }
    launch_pack_w(stream, conv_in_w, wt_in, WIDTH, 263, 3, CIN_PAD);
    for (int i = 0; i < 3; ++i)
        launch_pack_w(stream, down_w + (size_t)i * WIDTH * WIDTH * 4,
                      wt_down + (size_t)i * WIDTH * 4 * WIDTH, WIDTH, WIDTH, 4, WIDTH);
    for (int i = 0; i < 3; ++i)
        for (int d = 0; d < 3; ++d) {
            int s = i * 3 + d;
            launch_pack_w(stream, res_w1 + (size_t)s * WIDTH * WIDTH * 3,
                          wt_r1 + (size_t)s * WIDTH * 3 * WIDTH, WIDTH, WIDTH, 3, WIDTH);
            launch_pack_w(stream, res_w2 + (size_t)s * WIDTH * WIDTH * 1,
                          wt_r2 + (size_t)s * WIDTH * 1 * WIDTH, WIDTH, WIDTH, 1, WIDTH);
        }
    launch_pack_w(stream, conv_out_w, wt_out, CODE_DIM, WIDTH, 3, WIDTH);
    pack_cb_kernel<<<NB_CODE, 256, 0, stream>>>(codebook, cbh, cnorm);

    // ---- encoder ----
    // conv_in + relu: xp(T=2048,C=288) -> actA(T=2048,C=512)
    launch_conv(stream, xp, wt_in, conv_in_b, nullptr, actA,
                2048, 2048, CIN_PAD, WIDTH, 3, 1, 1, 1, 0, 1);

    bf16_t* hin = actA;     // current input, T = Tlev
    int T = 2048;
    bf16_t* bufs[2] = { actB, actA };   // down outputs alternate B, A, B
    for (int i = 0; i < 3; ++i) {
        bf16_t* hd = bufs[i & 1];
        int To = T / 2;
        // down conv: k=4, stride=2, pad=1
        launch_conv(stream, hin, wt_down + (size_t)i * WIDTH * 4 * WIDTH,
                    down_b + (size_t)i * WIDTH, nullptr, hd,
                    T, To, WIDTH, WIDTH, 4, 2, 1, 1, 0, 0);
        int dil = 1;
        for (int d = 0; d < 3; ++d) {
            int s = i * 3 + d;
            // r = relu(conv(relu(hd), w1, b1, pad=dil, dil=dil))
            launch_conv(stream, hd, wt_r1 + (size_t)s * WIDTH * 3 * WIDTH,
                        res_b1 + (size_t)s * WIDTH, nullptr, actC,
                        To, To, WIDTH, WIDTH, 3, 1, dil, dil, 1, 1);
            // hd = conv(r, w2, b2, k=1) + hd   (elementwise residual, safe in place)
            launch_conv(stream, actC, wt_r2 + (size_t)s * WIDTH * 1 * WIDTH,
                        res_b2 + (size_t)s * WIDTH, hd, hd,
                        To, To, WIDTH, WIDTH, 1, 1, 0, 1, 0, 0);
            dil *= 3;
        }
        hin = hd;
        T = To;
    }
    // conv_out: hin(T=256) -> e in actC region (8 MB, fits)
    bf16_t* e = actC;
    launch_conv(stream, hin, wt_out, conv_out_b, nullptr, e,
                256, 256, WIDTH, CODE_DIM, 3, 1, 1, 1, 0, 0);

    // ---- quantization ----
    vq_init_kernel<<<(NB_CODE + 255) / 256, 256, 0, stream>>>(counts, out_loss);
    vq_argmin_kernel<<<NROWS / 16, 32, 0, stream>>>(e, cbh, cnorm, codeidx);
    vq_count_kernel<<<(NROWS + 255) / 256, 256, 0, stream>>>(codeidx, counts);
    {
        size_t total = (size_t)NIMG * WIDTH * TQ;
        vq_gather_kernel<<<(unsigned)((total + 255) / 256), 256, 0, stream>>>(codeidx, codebook, out);
    }
    vq_commit_kernel<<<NROWS, 256, 0, stream>>>(e, codebook, codeidx, out_loss);
    vq_perplexity_kernel<<<1, 1024, 0, stream>>>(counts, out_perp);
}
